// DCU_52450140619544
// MI455X (gfx1250) — matrix operand order repaired and checked by reading, not compiled, not run on hardware
//
#include <hip/hip_runtime.h>
#include <hip/hip_bf16.h>
#include <math.h>

typedef __attribute__((ext_vector_type(16))) __bf16 v16bf;
typedef __attribute__((ext_vector_type(8)))  __bf16 v8bf;
typedef __attribute__((ext_vector_type(8)))  float  v8f;

// Problem dims (fixed by the reference)
static constexpr int B_    = 4;
static constexpr int CIN   = 128;
static constexpr int COUT  = 64;
static constexpr int Hh    = 64, Wh = 64;
static constexpr int H     = 128, W = 128;
static constexpr int HW    = H * W;
static constexpr int KTAP  = 9;
static constexpr int MOFF  = 27;
static constexpr int MOFFP = 32;
static constexpr int KOFF  = 128 * KTAP;   // 1152
static constexpr int KAL   = COUT * KTAP;  // 576

// Padded LDS row strides (elements). All are multiples of 8 (16B-aligned rows)
// and give odd-multiple-of-4 dword strides -> banks spread across fragment lanes.
static constexpr int LDA_OFF = 1160;  // A rows, offset conv  (2320 B = 580 dw)
static constexpr int LDA_AL  = 584;   // A rows, align conv   (1168 B = 292 dw)
static constexpr int LDB_AL  = 584;   // B rows [n][k], align conv
static constexpr int PADC    = 136;   // channels per (row,col) cell, offset patch

// ---------------------------------------------------------------------------
// WMMA fragment loaders. Fragment element e maps to:
//   A (16x32): e -> k0+kb+e                                            (kb=0|16)
//   B (32x16): e -> k0+kb+e                                            (kb=0|16)
// so fragments are 1-2 contiguous 16B runs -> ds_load_b128 pairs.
// ---------------------------------------------------------------------------
__device__ __forceinline__ v16bf frag_join(v8bf lo, v8bf hi) {
  return __builtin_shufflevector(lo, hi, 0, 1, 2, 3, 4, 5, 6, 7,
                                 8, 9, 10, 11, 12, 13, 14, 15);
}

__device__ __forceinline__ v16bf frag_a_ld(const __bf16* A, int lda, int k0, int lane) {
  const int m  = lane & 15;
  const int kb = (lane < 16) ? 0 : 16;
  const __bf16* p = A + m * lda + k0 + kb;
  return frag_join(*(const v8bf*)p, *(const v8bf*)(p + 8));
}

// B tile stored [N][K] with row stride ldb
__device__ __forceinline__ v16bf frag_b_ld(const __bf16* Bt, int ldb, int k0, int lane) {
  const int n  = lane & 15;
  const int kb = (lane < 16) ? 0 : 16;
  const __bf16* p = Bt + n * ldb + k0 + kb;
  return frag_join(*(const v8bf*)p, *(const v8bf*)(p + 8));
}

// ---------------------------------------------------------------------------
// K1: transposed conv 3x3 stride 2 -> up (fp32) + bf16 concat channels [0,64)
// ---------------------------------------------------------------------------
__global__ void k_convT(const float* __restrict__ high, const float* __restrict__ w_up,
                        const float* __restrict__ b_up, float* __restrict__ up,
                        __bf16* __restrict__ cat) {
  const int idx = blockIdx.x * blockDim.x + threadIdx.x;
  if (idx >= B_ * COUT * HW) return;
  const int x = idx & (W - 1);
  const int y = (idx >> 7) & (H - 1);
  const int o = (idx >> 14) & (COUT - 1);
  const int b = idx >> 20;

  int yi[2], wky[2], nty = 0;
  int xi[2], wkx[2], ntx = 0;
#pragma unroll
  for (int k = 0; k < 3; ++k) {
    const int t = y - 1 + k;
    if (!(t & 1)) { const int q = t >> 1; if (q >= 0 && q < Hh) { yi[nty] = q; wky[nty] = 2 - k; ++nty; } }
  }
#pragma unroll
  for (int k = 0; k < 3; ++k) {
    const int t = x - 1 + k;
    if (!(t & 1)) { const int q = t >> 1; if (q >= 0 && q < Wh) { xi[ntx] = q; wkx[ntx] = 2 - k; ++ntx; } }
  }

  float s = b_up[o];
  const float* hb = high + (size_t)b * CIN * Hh * Wh;
  for (int i = 0; i < CIN; ++i) {
    const float* hc = hb + i * Hh * Wh;
    const float* wc = w_up + (i * COUT + o) * 9;
    for (int ty = 0; ty < nty; ++ty)
      for (int tx = 0; tx < ntx; ++tx)
        s += hc[yi[ty] * Wh + xi[tx]] * wc[wky[ty] * 3 + wkx[tx]];
  }
  up[idx] = s;
  cat[((b * 2 * COUT + o) * H + y) * W + x] = (__bf16)s;
}

// ---------------------------------------------------------------------------
// K2: copy low -> out channels [64,128) (fp32) and concat channels (bf16)
// ---------------------------------------------------------------------------
__global__ void k_low(const float* __restrict__ low, float* __restrict__ out,
                      __bf16* __restrict__ cat) {
  const int idx = blockIdx.x * blockDim.x + threadIdx.x;
  if (idx >= B_ * COUT * HW) return;
  const int x = idx & (W - 1);
  const int y = (idx >> 7) & (H - 1);
  const int c = (idx >> 14) & (COUT - 1);
  const int b = idx >> 20;
  const float v = low[idx];
  out[((b * 128 + 64 + c) * H + y) * W + x] = v;
  cat[((b * 128 + 64 + c) * H + y) * W + x] = (__bf16)v;
}

// ---------------------------------------------------------------------------
// K3/K4: repack weights to bf16 padded [M][K'] (k = tap*C + c, zero padding)
// ---------------------------------------------------------------------------
__global__ void k_pack_woff(const float* __restrict__ w_off, __bf16* __restrict__ dst) {
  const int i = blockIdx.x * blockDim.x + threadIdx.x;
  if (i >= MOFFP * LDA_OFF) return;
  const int m = i / LDA_OFF, k = i % LDA_OFF;
  __bf16 v = (__bf16)0.f;
  if (m < MOFF && k < KOFF) {
    const int tap = k / 128, c = k % 128;
    v = (__bf16)w_off[(m * 128 + c) * 9 + tap];
  }
  dst[i] = v;
}

__global__ void k_pack_walign(const float* __restrict__ w_al, __bf16* __restrict__ dst) {
  const int i = blockIdx.x * blockDim.x + threadIdx.x;
  if (i >= COUT * LDA_AL) return;
  const int o = i / LDA_AL, k = i % LDA_AL;
  __bf16 v = (__bf16)0.f;
  if (k < KAL) {
    const int kk = k / COUT, c = k % COUT;
    v = (__bf16)w_al[(o * COUT + c) * 9 + kk];
  }
  dst[i] = v;
}

// ---------------------------------------------------------------------------
// K5: offset conv (3x3, pad 1) implicit GEMM, M=32(27), K=1152, WMMA bf16
//     grid: b*256 + y*2 + xblk (64 pixels); 256 threads = 8 waves
//     LDS: sA [32][1160] + patch sP [3][66][136] (channels contiguous)
// ---------------------------------------------------------------------------
static constexpr int P_COLS  = 66;
static constexpr int SZ_WA   = MOFFP * LDA_OFF;       // 37120 elems
static constexpr int SZ_PT   = 3 * P_COLS * PADC;     // 26928 elems
static constexpr int N_PATCH = 3 * P_COLS * 128;      // real elements staged

__global__ void k_offconv(const __bf16* __restrict__ cat, const __bf16* __restrict__ woff,
                          const float* __restrict__ b_off, float* __restrict__ fo) {
  extern __shared__ __bf16 smem[];
  __bf16* sA = smem;            // [32][LDA_OFF]
  __bf16* sP = smem + SZ_WA;    // [(r*66+col)][PADC]

  const int tid  = threadIdx.x;
  const int xblk = blockIdx.x & 1;
  const int y    = (blockIdx.x >> 1) & (H - 1);
  const int b    = blockIdx.x >> 8;
  const int x0   = xblk * 64;

  for (int i = tid; i < SZ_WA; i += 256) sA[i] = woff[i];

  // Stage patch: i -> (c, r, col) with col innermost (coalesced global reads);
  // LDS writes stride PADC*2=272B per col -> banks spread (68 dw, odd multiple of 4).
  for (int i = tid; i < N_PATCH; i += 256) {
    const int col = i % P_COLS;
    const int r   = (i / P_COLS) % 3;
    const int c   = i / (3 * P_COLS);
    const int yy  = y - 1 + r;
    const int xx  = x0 - 1 + col;
    __bf16 v = (__bf16)0.f;
    if (yy >= 0 && yy < H && xx >= 0 && xx < W)
      v = cat[((b * 128 + c) * H + yy) * W + xx];
    sP[(r * P_COLS + col) * PADC + c] = v;
  }
  __syncthreads();

  const int lane  = tid & 31;
  const int wave  = tid >> 5;
  const int mtile = wave & 1;
  const int ntile = wave >> 1;   // 0..3

  const int n  = lane & 15;
  const int kb = (lane < 16) ? 0 : 16;
  const __bf16* aRow = sA + mtile * 16 * LDA_OFF;

  v8f acc = {};
#pragma unroll 3
  for (int tap = 0; tap < 9; ++tap) {
    const int ky = tap / 3, kx = tap % 3;
    // fragment channel-run base for this lane's column
    const __bf16* pcell = sP + (ky * P_COLS + ntile * 16 + kx + n) * PADC + kb;
#pragma unroll
    for (int cb = 0; cb < 128; cb += 32) {
      const v16bf fa = frag_a_ld(aRow, LDA_OFF, tap * 128 + cb, lane);
      const __bf16* p = pcell + cb;
      const v16bf fb = frag_join(*(const v8bf*)p, *(const v8bf*)(p + 8));
      acc = __builtin_amdgcn_wmma_f32_16x16x32_bf16(false, fa, false, fb,
                                                    (short)0, acc, false, false);
    }
  }

  const int x = x0 + ntile * 16 + n;
#pragma unroll
  for (int v = 0; v < 8; ++v) {
    const int m = mtile * 16 + ((lane < 16) ? v : v + 8);
    if (m < MOFF)
      fo[((b * MOFF + m) * H + y) * W + x] = acc[v] + b_off[m];
  }
}

// ---------------------------------------------------------------------------
// K6: modulated deformable conv + bias + LeakyReLU(0.2), WMMA einsum
//     grid: b*1024 + y*8 + xblk (16 pixels); 128 threads = 4 waves
//     LDS: sW [64][584] + sV [16][584] ([n][k], k contiguous)
// ---------------------------------------------------------------------------
static constexpr int SZ_WAL = COUT * LDA_AL;   // 37376 elems
static constexpr int SZ_VT  = 16 * LDB_AL;     // 9344 elems

__global__ void k_deform(const float* __restrict__ up, const float* __restrict__ fo,
                         const __bf16* __restrict__ wal, const float* __restrict__ b_al,
                         float* __restrict__ out) {
  extern __shared__ __bf16 smem[];
  __bf16* sW = smem;            // [64][LDA_AL]
  __bf16* sV = smem + SZ_WAL;   // [16][LDB_AL]

  const int tid  = threadIdx.x;
  const int xblk = blockIdx.x & 7;
  const int y    = (blockIdx.x >> 3) & (H - 1);
  const int b    = blockIdx.x >> 10;
  const int x0   = xblk * 16;

  for (int i = tid; i < SZ_WAL; i += 128) sW[i] = wal[i];

  // Build B tile [n][k], k = kk*64 + c contiguous per (n,kk) pair
  const float* upb = up + (size_t)b * COUT * HW;
  for (int p = tid; p < 16 * KTAP; p += 128) {
    const int n  = p & 15;
    const int kk = p >> 4;
    const int x  = x0 + n;
    const float oy = fo[((b * MOFF + 2 * kk)     * H + y) * W + x];
    const float ox = fo[((b * MOFF + 2 * kk + 1) * H + y) * W + x];
    const float mz = fo[((b * MOFF + 18 + kk)    * H + y) * W + x];
    const float mk = 2.f / (1.f + __expf(-mz));
    const int ky = kk / 3, kx = kk % 3;
    const float py = oy + (float)(y - 1 + ky);
    const float px = ox + (float)(x - 1 + kx);
    const float fy = floorf(py), fx = floorf(px);
    const float wy = py - fy,    wx = px - fx;
    const int y0 = (int)fy, x0i = (int)fx;
    const int y1 = y0 + 1,  x1  = x0i + 1;
    const bool vy0 = (y0 >= 0 && y0 < H), vy1 = (y1 >= 0 && y1 < H);
    const bool vx0 = (x0i >= 0 && x0i < W), vx1 = (x1 >= 0 && x1 < W);
    const int cy0 = vy0 ? y0 : (y0 < 0 ? 0 : H - 1);
    const int cy1 = vy1 ? y1 : (y1 < 0 ? 0 : H - 1);
    const int cx0 = vx0 ? x0i : (x0i < 0 ? 0 : W - 1);
    const int cx1 = vx1 ? x1 : (x1 < 0 ? 0 : W - 1);
    const float w00 = (vy0 && vx0) ? (1.f - wy) * (1.f - wx) : 0.f;
    const float w01 = (vy0 && vx1) ? (1.f - wy) * wx         : 0.f;
    const float w10 = (vy1 && vx0) ? wy * (1.f - wx)         : 0.f;
    const float w11 = (vy1 && vx1) ? wy * wx                 : 0.f;
    __bf16* dst = sV + n * LDB_AL + kk * COUT;
    for (int c = 0; c < COUT; ++c) {
      const float* ch = upb + c * HW;
      const float g = w00 * ch[cy0 * W + cx0] + w01 * ch[cy0 * W + cx1] +
                      w10 * ch[cy1 * W + cx0] + w11 * ch[cy1 * W + cx1];
      dst[c] = (__bf16)(mk * g);
    }
  }
  __syncthreads();

  const int lane  = tid & 31;
  const int mtile = tid >> 5;   // 0..3

  v8f acc = {};
#pragma unroll
  for (int ks = 0; ks < KAL / 32; ++ks) {
    const v16bf fa = frag_a_ld(sW + mtile * 16 * LDA_AL, LDA_AL, ks * 32, lane);
    const v16bf fb = frag_b_ld(sV, LDB_AL, ks * 32, lane);
    acc = __builtin_amdgcn_wmma_f32_16x16x32_bf16(false, fa, false, fb,
                                                  (short)0, acc, false, false);
  }

  const int x = x0 + (lane & 15);
#pragma unroll
  for (int v = 0; v < 8; ++v) {
    const int m = mtile * 16 + ((lane < 16) ? v : v + 8);
    float d = acc[v] + b_al[m];
    d = (d >= 0.f) ? d : 0.2f * d;
    out[((b * 128 + m) * H + y) * W + x] = d;
  }
}

// ---------------------------------------------------------------------------
// Launch
// ---------------------------------------------------------------------------
extern "C" void kernel_launch(void* const* d_in, const int* in_sizes, int n_in,
                              void* d_out, int out_size, void* d_ws, size_t ws_size,
                              hipStream_t stream) {
  const float* high  = (const float*)d_in[0];
  const float* low   = (const float*)d_in[1];
  const float* w_up  = (const float*)d_in[2];
  const float* b_up  = (const float*)d_in[3];
  const float* w_off = (const float*)d_in[4];
  const float* b_off = (const float*)d_in[5];
  const float* w_al  = (const float*)d_in[6];
  const float* b_al  = (const float*)d_in[7];
  float* out = (float*)d_out;

  char* ws = (char*)d_ws;
  float*  up   = (float*)(ws);                                   // 16,777,216 B
  __bf16* cat  = (__bf16*)(ws + 16777216);                       // 16,777,216 B
  float*  fo   = (float*)(ws + 33554432);                        //  7,077,888 B
  __bf16* woff = (__bf16*)(ws + 40632320);                       //     74,240 B
  __bf16* wal  = (__bf16*)(ws + 40706560);                       //     74,752 B

  const int nUp = B_ * COUT * HW;
  k_convT<<<(nUp + 255) / 256, 256, 0, stream>>>(high, w_up, b_up, up, cat);
  k_low  <<<(nUp + 255) / 256, 256, 0, stream>>>(low, out, cat);
  k_pack_woff  <<<(MOFFP * LDA_OFF + 255) / 256, 256, 0, stream>>>(w_off, woff);
  k_pack_walign<<<(COUT * LDA_AL  + 255) / 256, 256, 0, stream>>>(w_al, wal);

  const size_t lds_off = (size_t)(SZ_WA + SZ_PT) * sizeof(__bf16);   // 128,096 B
  k_offconv<<<B_ * H * 2, 256, lds_off, stream>>>(cat, woff, b_off, fo);

  const size_t lds_def = (size_t)(SZ_WAL + SZ_VT) * sizeof(__bf16);  //  93,440 B
  k_deform<<<B_ * H * 8, 128, lds_def, stream>>>(up, fo, wal, b_al, out);
}
